// TreeNode_42417097015806
// MI455X (gfx1250) — matrix operand order repaired and checked by reading, not compiled, not run on hardware
//
#include <hip/hip_runtime.h>

typedef __attribute__((ext_vector_type(16))) __bf16 v16bf;
typedef __attribute__((ext_vector_type(8)))  float  v8f;

#define TD 256   // feature dim D
#define TC 64    // num classes C

// ---------------------------------------------------------------------------
// Pack w_left / w_right (row-major [D, C] fp32) into bf16 WMMA B-fragments.
// B (32x16, bf16) lane layout: lanes 0-15 hold K=0..7,16..23 (2 K per VGPR),
// lanes 16-31 hold K=8..15,24..31; lane&15 selects the N column.
// Flat index: (((mat*8 + kb)*4 + nt)*32 + lane)*16 + i  with  K = kb*32 +
// ((lane&16)?8:0) + i + (i&8),  col = nt*16 + (lane&15).
// One thread per (mat, kb, nt, lane) = 2048 threads, 16 bf16 each.
// ---------------------------------------------------------------------------
__global__ __launch_bounds__(256) void pack_weights_kernel(
    const float* __restrict__ wl, const float* __restrict__ wr,
    __bf16* __restrict__ packed)
{
  int tid = blockIdx.x * 256 + threadIdx.x;
  if (tid >= 2048) return;
  int lane = tid & 31;
  int nt   = (tid >> 5) & 3;
  int kb   = (tid >> 7) & 7;
  int mat  = (tid >> 10) & 1;
  const float* __restrict__ w = mat ? wr : wl;
  int kbase = kb * 32 + ((lane & 16) ? 8 : 0);
  int c = nt * 16 + (lane & 15);
  __bf16* __restrict__ dst = packed + (long long)tid * 16;
#pragma unroll
  for (int i = 0; i < 16; ++i)
    dst[i] = (__bf16)w[(kbase + i + (i & 8)) * TC + c];
}

// ---------------------------------------------------------------------------
// Main kernel: 8 waves per WG; each wave owns TWO 16-row tiles (32 rows), so
// every B fragment read from LDS feeds two v_wmma_f32_16x16x32_bf16 ops
// (halves DS bytes per FLOP). Accumulators start at the bias so the epilogue
// is a pure router-masked select.
// ---------------------------------------------------------------------------
__global__ __launch_bounds__(256, 1) void tree_route_kernel(
    const float*  __restrict__ x,
    const float*  __restrict__ w_router,
    const float*  __restrict__ b_router,
    const float*  __restrict__ b_left,
    const float*  __restrict__ b_right,
    const __bf16* __restrict__ packedW,
    float*        __restrict__ out,
    int nrows)
{
  __shared__ __align__(32) __bf16 sW[2 * 8 * 4 * 32 * 16];   // 64 KB packed weights
  __shared__ float sWr[TD];
  __shared__ float sBias[2 * TC];

  const int t = threadIdx.x;
  {
    const uint4* __restrict__ src = (const uint4*)packedW;
    uint4* dst = (uint4*)sW;                                  // 4096 uint4 total
#pragma unroll
    for (int i = 0; i < 16; ++i) dst[t + i * 256] = src[t + i * 256];
    sWr[t] = w_router[t];                                     // 256 threads == TD
    if (t < TC) { sBias[t] = b_left[t]; sBias[TC + t] = b_right[t]; }
  }
  __syncthreads();

  const int wave = t >> 5;
  const int lane = t & 31;
  const long long rowBase = ((long long)blockIdx.x * 8 + wave) * 32;
  if (rowBase + 32 > nrows) return;

  // A-fragment addressing (16-bit A 16x32): row m = lane&15; lanes>=16 take
  // the +8 K offset in each 16-wide half.
  const int m = lane & 15;
  const int halfK = (lane & 16) ? 8 : 0;
  const float* __restrict__ xrow0 = x + (rowBase + m) * TD;        // tile 0
  const float* __restrict__ xrow1 = x + (rowBase + 16 + m) * TD;   // tile 1
  const float br = b_router[0];
  const int c15 = lane & 15;

  // acc[tile*8 + child*4 + nt]; seed with bias (uniform over a lane's 8 rows).
  v8f acc[16];
#pragma unroll
  for (int nt = 0; nt < 4; ++nt) {
    const float bl = sBias[nt * 16 + c15];
    const float brt = sBias[TC + nt * 16 + c15];
    v8f vl, vr;
#pragma unroll
    for (int v = 0; v < 8; ++v) { vl[v] = bl; vr[v] = brt; }
    acc[0 + nt] = vl;  acc[4 + nt] = vr;    // tile 0: left, right
    acc[8 + nt] = vl;  acc[12 + nt] = vr;   // tile 1: left, right
  }

  float rsum0 = 0.0f, rsum1 = 0.0f;   // fp32 router partial dots

#pragma unroll
  for (int kb = 0; kb < 8; ++kb) {
    const int k0 = kb * 32 + halfK;

    // ---- A fragment, tile 0 ----
    const float4 a0 = *(const float4*)(xrow0 + k0);
    const float4 a1 = *(const float4*)(xrow0 + k0 + 4);
    const float4 a2 = *(const float4*)(xrow0 + k0 + 16);
    const float4 a3 = *(const float4*)(xrow0 + k0 + 20);
    const float xv0[16] = {a0.x, a0.y, a0.z, a0.w, a1.x, a1.y, a1.z, a1.w,
                           a2.x, a2.y, a2.z, a2.w, a3.x, a3.y, a3.z, a3.w};
    // ---- A fragment, tile 1 ----
    const float4 b0 = *(const float4*)(xrow1 + k0);
    const float4 b1 = *(const float4*)(xrow1 + k0 + 4);
    const float4 b2 = *(const float4*)(xrow1 + k0 + 16);
    const float4 b3 = *(const float4*)(xrow1 + k0 + 20);
    const float xv1[16] = {b0.x, b0.y, b0.z, b0.w, b1.x, b1.y, b1.z, b1.w,
                           b2.x, b2.y, b2.z, b2.w, b3.x, b3.y, b3.z, b3.w};

    v16bf fa0, fa1;
#pragma unroll
    for (int i = 0; i < 16; ++i) {
      fa0[i] = (__bf16)xv0[i];
      fa1[i] = (__bf16)xv1[i];
      const float wr = sWr[k0 + i + (i & 8)];   // element i -> K = k0+i (+8 hi)
      rsum0 = __builtin_fmaf(xv0[i], wr, rsum0);
      rsum1 = __builtin_fmaf(xv1[i], wr, rsum1);
    }

#pragma unroll
    for (int nt = 0; nt < 4; ++nt) {
      const v16bf bL = *(const v16bf*)(sW + ((((0 * 8 + kb) * 4 + nt) * 32 + lane) * 16));
      acc[0 + nt] = __builtin_amdgcn_wmma_f32_16x16x32_bf16(
          false, fa0, false, bL, (short)0, acc[0 + nt], false, false);
      acc[8 + nt] = __builtin_amdgcn_wmma_f32_16x16x32_bf16(
          false, fa1, false, bL, (short)0, acc[8 + nt], false, false);
      const v16bf bR = *(const v16bf*)(sW + ((((1 * 8 + kb) * 4 + nt) * 32 + lane) * 16));
      acc[4 + nt] = __builtin_amdgcn_wmma_f32_16x16x32_bf16(
          false, fa0, false, bR, (short)0, acc[4 + nt], false, false);
      acc[12 + nt] = __builtin_amdgcn_wmma_f32_16x16x32_bf16(
          false, fa1, false, bR, (short)0, acc[12 + nt], false, false);
    }
  }

  // Combine K-halves of the router dots (lanes l and l+16 share row m).
  rsum0 += __shfl_xor(rsum0, 16, 32);
  rsum1 += __shfl_xor(rsum1, 16, 32);
  // sigmoid(t) > 0.5  <=>  t > 0 ; bits 0..15 of the wave32 ballot = row masks.
  const unsigned mask0 = (unsigned)__ballot((rsum0 + br) > 0.0f) & 0xFFFFu;
  const unsigned mask1 = (unsigned)__ballot((rsum1 + br) > 0.0f) & 0xFFFFu;

  // C/D layout: VGPR v -> row v (lanes 0-15) or row v+8 (lanes 16-31),
  // column = lane&15 within each 16-wide N tile.
  const int rb = (lane & 16) ? 8 : 0;
#pragma unroll
  for (int tile = 0; tile < 2; ++tile) {
    float* __restrict__ orow = out + (rowBase + tile * 16) * TC;
    const unsigned mask = tile ? mask1 : mask0;
    const int abase = tile * 8;
#pragma unroll
    for (int v = 0; v < 8; ++v) {
      const int row = rb + v;
      const bool left = (mask >> row) & 1u;
#pragma unroll
      for (int nt = 0; nt < 4; ++nt) {
        const int c = nt * 16 + c15;
        orow[(long long)row * TC + c] =
            left ? acc[abase + nt][v] : acc[abase + 4 + nt][v];
      }
    }
  }
}

// ---------------------------------------------------------------------------
// Inputs (setup_inputs order): x[N*256], w_router[256], b_router[1],
// w_left[256*64], b_left[64], w_right[256*64], b_right[64].
// Output: float[N*64].
// ---------------------------------------------------------------------------
extern "C" void kernel_launch(void* const* d_in, const int* in_sizes, int n_in,
                              void* d_out, int out_size, void* d_ws, size_t ws_size,
                              hipStream_t stream) {
  const float* x        = (const float*)d_in[0];
  const float* w_router = (const float*)d_in[1];
  const float* b_router = (const float*)d_in[2];
  const float* w_left   = (const float*)d_in[3];
  const float* b_left   = (const float*)d_in[4];
  const float* w_right  = (const float*)d_in[5];
  const float* b_right  = (const float*)d_in[6];
  float* out = (float*)d_out;
  __bf16* packed = (__bf16*)d_ws;   // 32768 bf16 = 64 KB of scratch

  const long long nrows = (long long)in_sizes[0] / TD;

  pack_weights_kernel<<<8, 256, 0, stream>>>(w_left, w_right, packed);

  const int nwg = (int)((nrows + 255) / 256);   // 256 rows per WG (8 waves x 32)
  tree_route_kernel<<<nwg, 256, 0, stream>>>(x, w_router, b_router, b_left,
                                             b_right, packed, out, (int)nrows);
}
